// STSGCL_67164698574916
// MI455X (gfx1250) — matrix operand order repaired and checked by reading, not compiled, not run on hardware
//
#include <hip/hip_runtime.h>
#include <hip/hip_bf16.h>
#include <cstdint>
#include <cstddef>

typedef __bf16 bf16_t;
typedef __attribute__((ext_vector_type(16))) __bf16 v16bf;
typedef __attribute__((ext_vector_type(8)))  float  v8f;

// ---- CDNA5 async global->LDS copy (gate on builtin availability) ----
#if defined(__has_builtin)
#if __has_builtin(__builtin_amdgcn_global_load_async_to_lds_b128) && \
    __has_builtin(__builtin_amdgcn_s_wait_asynccnt)
#define USE_ASYNC_LDS 1
#endif
#endif
#ifndef USE_ASYNC_LDS
#define USE_ASYNC_LDS 0
#endif

__device__ inline void cp_async16(bf16_t* lds_dst, const bf16_t* gsrc) {
#if USE_ASYNC_LDS
  typedef int v4i __attribute__((vector_size(16)));
  typedef __attribute__((address_space(1))) v4i* gv4ip;
  typedef __attribute__((address_space(3))) v4i* lv4ip;
  __builtin_amdgcn_global_load_async_to_lds_b128(
      (gv4ip)(uintptr_t)gsrc,
      (lv4ip)(uint32_t)(uintptr_t)lds_dst,
      0, 0);
#else
  (void)lds_dst; (void)gsrc;
#endif
}

// ---- problem dims ----
constexpr int B_  = 32, C_ = 64, T_ = 12, N_ = 883;
constexpr int TM2 = 10;                    // T-2
constexpr int NP  = 888;                   // padded vertex block => 16B-aligned columns
constexpr int KP  = 2688;                  // padded K (3*NP=2664 -> 84*32)
constexpr int KSTEPS = KP / 32;            // 84
constexpr int MP  = 896;                   // padded middle-row count (883 -> 56*16)
constexpr int BC  = B_ * C_;               // 2048
constexpr int PTOT = BC * TM2;             // 20480 gemm1 columns
constexpr long HTOT = (long)BC * T_ * NP;  // 21,823,488
constexpr long HPAD = HTOT + 64;           // overread pad for last column

// ---- workspace layout (bytes) ----
constexpr size_t OFF_HB  = 0;
constexpr size_t SZ_HB   = (size_t)HPAD * 2;
constexpr size_t OFF_ADJ = (OFF_HB + SZ_HB + 255) & ~(size_t)255;
constexpr size_t SZ_ADJ  = (size_t)MP * KP * 2;
constexpr size_t OFF_WT  = (OFF_ADJ + SZ_ADJ + 255) & ~(size_t)255;
constexpr size_t SZ_WT   = (size_t)3 * 128 * 64 * 2;
constexpr size_t OFF_G   = (OFF_WT + SZ_WT + 255) & ~(size_t)255;   // [B*TM2][MP][C] bf16

// ============ prep kernels ============

__global__ __launch_bounds__(256) void prep_h(const float* __restrict__ x,
                                              const float* __restrict__ temb,
                                              const float* __restrict__ semb,
                                              bf16_t* __restrict__ hb) {
  const long i = (long)blockIdx.x * 256 + threadIdx.x;
  if (i >= HPAD) return;
  float v = 0.f;
  if (i < HTOT) {
    const int  vv  = (int)(i % NP);
    const long rem = i / NP;
    const int  t   = (int)(rem % T_);
    const int  bc  = (int)(rem / T_);
    if (vv < N_) {
      const int c = bc & 63;
      v = x[((size_t)bc * T_ + t) * N_ + vv] + temb[c * T_ + t] + semb[c * N_ + vv];
    }
  }
  hb[i] = (bf16_t)v;
}

__global__ __launch_bounds__(256) void prep_adj(const float* __restrict__ adj,
                                                bf16_t* __restrict__ adjm) {
  const long i = (long)blockIdx.x * 256 + threadIdx.x;
  if (i >= (long)MP * KP) return;
  const int k = (int)(i % KP);
  const int m = (int)(i / KP);
  const int j = k / NP, vv = k % NP;
  float v = 0.f;
  if (m < N_ && j < 3 && vv < N_)
    v = adj[(size_t)(N_ + m) * (3 * N_) + j * N_ + vv];  // middle-block rows only
  adjm[i] = (bf16_t)v;
}

__global__ __launch_bounds__(256) void prep_w(const float* __restrict__ W,
                                              bf16_t* __restrict__ wt) {
  const int i = blockIdx.x * 256 + threadIdx.x;
  if (i >= 3 * 128 * 64) return;
  const int c  = i & 63;
  const int d  = (i >> 6) & 127;
  const int gi = i >> 13;
  wt[i] = (bf16_t)W[gi * 8192 + c * 128 + d];            // Wt[gi][d][c]
}

// ============ GEMM1: g[m, (b,c,t)] = adjm[m,:] . h[b,c,t, 0:3N] ============

__global__ __launch_bounds__(256) void gemm1_adj(const bf16_t* __restrict__ adjm,
                                                 const bf16_t* __restrict__ hb,
                                                 bf16_t* __restrict__ gout) {
  __shared__ bf16_t As[2][128][48];   // [m_local][k], stride 48 keeps 16B align
  __shared__ bf16_t Bs[2][128][48];   // [p_local][k]

  const int mt = blockIdx.x % 7;
  const int pt = blockIdx.x / 7;
  const int mbase = mt * 128;
  const int pbase = pt * 128;

  const int tid  = threadIdx.x;
  // staging role: 2 threads per row/column, 16 bf16 (32B) each
  const int srow = tid >> 1;
  const int koff = (tid & 1) << 4;
  const int pg   = pbase + srow;
  const int bc_s = pg / 10;
  const int tt_s = pg - bc_s * 10;
  const bf16_t* aptr = adjm + (size_t)(mbase + srow) * KP + koff;
  const bf16_t* bptr = hb + (size_t)(bc_s * 12 + tt_s) * NP + koff;  // contiguous 3-window column

#if USE_ASYNC_LDS
  auto issue = [&](int kb, int buf) {
    cp_async16(&As[buf][srow][koff],     aptr + kb);
    cp_async16(&As[buf][srow][koff + 8], aptr + kb + 8);
    cp_async16(&Bs[buf][srow][koff],     bptr + kb);
    cp_async16(&Bs[buf][srow][koff + 8], bptr + kb + 8);
  };
#else
  uint4 ra0, ra1, rb0, rb1;
  auto fetch = [&](int kb) {
    const bf16_t* a = aptr + kb;
    const bf16_t* b = bptr + kb;
    ra0 = *(const uint4*)(a);     ra1 = *(const uint4*)(a + 8);
    rb0 = *(const uint4*)(b);     rb1 = *(const uint4*)(b + 8);
  };
  auto stage = [&](int buf) {
    *(uint4*)&As[buf][srow][koff]     = ra0;
    *(uint4*)&As[buf][srow][koff + 8] = ra1;
    *(uint4*)&Bs[buf][srow][koff]     = rb0;
    *(uint4*)&Bs[buf][srow][koff + 8] = rb1;
  };
#endif

  const int w    = tid >> 5;
  const int lane = tid & 31;
  const int ln   = lane & 15;
  const int half = lane >> 4;
  const int wm   = w & 3;    // 4 row-groups of 32 (m)
  const int wp   = w >> 2;   // 2 col-groups of 64 (p)

  const v8f vzero = {0.f, 0.f, 0.f, 0.f, 0.f, 0.f, 0.f, 0.f};
  v8f acc[2][4];
#pragma unroll
  for (int a = 0; a < 2; ++a)
#pragma unroll
    for (int q = 0; q < 4; ++q) acc[a][q] = vzero;

  // compute stage (reads LDS buffer `cur`)
  auto compute = [&](int cur) {
    v16bf af[2];
#pragma unroll
    for (int a = 0; a < 2; ++a) {
      const bf16_t* r = &As[cur][wm * 32 + a * 16 + ln][0];
      uint4* fp = (uint4*)&af[a];
      fp[0] = *(const uint4*)(r + half * 8);
      fp[1] = *(const uint4*)(r + 16 + half * 8);
    }
    v16bf bfr[4];
#pragma unroll
    for (int q = 0; q < 4; ++q) {
      const bf16_t* r = &Bs[cur][wp * 64 + q * 16 + ln][0];
      uint4* fp = (uint4*)&bfr[q];
      fp[0] = *(const uint4*)(r + half * 8);
      fp[1] = *(const uint4*)(r + 16 + half * 8);
    }
#pragma unroll
    for (int a = 0; a < 2; ++a)
#pragma unroll
      for (int q = 0; q < 4; ++q)
        acc[a][q] = __builtin_amdgcn_wmma_f32_16x16x32_bf16(
            false, af[a], false, bfr[q], (short)0, acc[a][q], false, false);
  };

#if USE_ASYNC_LDS
  issue(0, 0);
  __builtin_amdgcn_s_wait_asynccnt(0);
  __syncthreads();
  for (int ks = 0; ks < KSTEPS; ++ks) {
    const int cur = ks & 1;
    if (ks + 1 < KSTEPS) issue((ks + 1) * 32, cur ^ 1);  // overlaps with compute
    compute(cur);
    if (ks + 1 < KSTEPS) __builtin_amdgcn_s_wait_asynccnt(0);
    __syncthreads();
  }
#else
  fetch(0);
  stage(0);
  __syncthreads();
  for (int ks = 0; ks < KSTEPS; ++ks) {
    const int cur = ks & 1;
    if (ks + 1 < KSTEPS) fetch((ks + 1) * 32);
    compute(cur);
    if (ks + 1 < KSTEPS) stage(cur ^ 1);
    __syncthreads();
  }
#endif

  // epilogue: g[(b*10+t)][m][c] in bf16 (contiguous-in-c for GEMM2's B operand)
#pragma unroll
  for (int q = 0; q < 4; ++q) {
    const int p_global = pbase + wp * 64 + q * 16 + ln;
    const int bc = p_global / 10;
    const int tt = p_global - bc * 10;
    const int bq = bc >> 6, cq = bc & 63;
    const size_t gb = ((size_t)(bq * 10 + tt) * MP) * 64 + cq;
#pragma unroll
    for (int a = 0; a < 2; ++a) {
      const int mrow = mbase + wm * 32 + a * 16 + half * 8;  // D row = r + 8*half
      v8f v = acc[a][q];
#pragma unroll
      for (int r = 0; r < 8; ++r)
        gout[gb + (size_t)(mrow + r) * 64] = (bf16_t)v[r];
    }
  }
}

// ============ GEMM2 + GLU + branch-max (fused) ============

__global__ __launch_bounds__(256) void gemm2_glu(const bf16_t* __restrict__ g,
                                                 const bf16_t* __restrict__ wt,
                                                 const float* __restrict__ bias,
                                                 float* __restrict__ out) {
  const int bt = blockIdx.x / 14;
  const int mc = blockIdx.x % 14;
  const int bq = bt / 10, tt = bt - bq * 10;

  const int tid  = threadIdx.x;
  const int w    = tid >> 5, lane = tid & 31, ln = lane & 15, half = lane >> 4;
  const int pair = w & 3;   // d-tile pair (value: pair, gate: pair+4)
  const int mh   = w >> 2;  // which 32-col half of the 64-wide m chunk

  const bf16_t* gb = g + (size_t)bt * MP * 64;

  const v8f vzero = {0.f, 0.f, 0.f, 0.f, 0.f, 0.f, 0.f, 0.f};
  v8f acc[3][2][2];
#pragma unroll
  for (int gi = 0; gi < 3; ++gi)
#pragma unroll
    for (int dh = 0; dh < 2; ++dh)
#pragma unroll
      for (int a = 0; a < 2; ++a) acc[gi][dh][a] = vzero;

#pragma unroll
  for (int ksv = 0; ksv < 2; ++ksv) {
    const int kb = ksv * 32;
    v16bf bfr[2];
#pragma unroll
    for (int a = 0; a < 2; ++a) {
      const int m = mc * 64 + mh * 32 + a * 16 + ln;
      const bf16_t* p = gb + (size_t)m * 64 + kb;   // contiguous in c
      uint4* fp = (uint4*)&bfr[a];
      fp[0] = *(const uint4*)(p + half * 8);
      fp[1] = *(const uint4*)(p + 16 + half * 8);
    }
#pragma unroll
    for (int gi = 0; gi < 3; ++gi)
#pragma unroll
      for (int dh = 0; dh < 2; ++dh) {
        const int d = (pair + dh * 4) * 16 + ln;
        const bf16_t* ap = wt + gi * 8192 + (size_t)d * 64 + kb;
        v16bf af;
        uint4* fp = (uint4*)&af;
        fp[0] = *(const uint4*)(ap + half * 8);
        fp[1] = *(const uint4*)(ap + 16 + half * 8);
#pragma unroll
        for (int a = 0; a < 2; ++a)
          acc[gi][dh][a] = __builtin_amdgcn_wmma_f32_16x16x32_bf16(
              false, af, false, bfr[a], (short)0, acc[gi][dh][a], false, false);
      }
  }

  // GLU + max over branches, write f32 output [B,C,10,N]
  const int dbase = pair * 16 + half * 8;   // c_out for element r is dbase + r
#pragma unroll
  for (int a = 0; a < 2; ++a) {
    const int mg = mc * 64 + mh * 32 + a * 16 + ln;
    float best[8];
#pragma unroll
    for (int gi = 0; gi < 3; ++gi) {
      v8f zv = acc[gi][0][a];
      v8f zg = acc[gi][1][a];
#pragma unroll
      for (int r = 0; r < 8; ++r) {
        const float v  = zv[r] + bias[gi * 128 + dbase + r];
        const float gx = zg[r] + bias[gi * 128 + 64 + dbase + r];
        const float o  = v * (1.0f / (1.0f + __expf(-gx)));
        best[r] = (gi == 0) ? o : fmaxf(best[r], o);
      }
    }
    if (mg < N_) {
#pragma unroll
      for (int r = 0; r < 8; ++r)
        out[((size_t)(bq * 64 + dbase + r) * 10 + tt) * N_ + mg] = best[r];
    }
  }
}

// ============ launch ============

extern "C" void kernel_launch(void* const* d_in, const int* in_sizes, int n_in,
                              void* d_out, int out_size, void* d_ws, size_t ws_size,
                              hipStream_t stream) {
  (void)in_sizes; (void)n_in; (void)out_size; (void)ws_size;
  const float* x    = (const float*)d_in[0];
  const float* temb = (const float*)d_in[1];
  const float* semb = (const float*)d_in[2];
  const float* adj  = (const float*)d_in[3];
  const float* W    = (const float*)d_in[4];
  const float* bias = (const float*)d_in[5];
  float* out = (float*)d_out;

  char* ws = (char*)d_ws;
  bf16_t* hb   = (bf16_t*)(ws + OFF_HB);
  bf16_t* adjm = (bf16_t*)(ws + OFF_ADJ);
  bf16_t* wt   = (bf16_t*)(ws + OFF_WT);
  bf16_t* g    = (bf16_t*)(ws + OFF_G);

  prep_h  <<<(unsigned)((HPAD + 255) / 256),            256, 0, stream>>>(x, temb, semb, hb);
  prep_adj<<<(unsigned)(((long)MP * KP + 255) / 256),   256, 0, stream>>>(adj, adjm);
  prep_w  <<<(3 * 128 * 64 + 255) / 256,                256, 0, stream>>>(W, wt);
  gemm1_adj<<<7 * (PTOT / 128),                         256, 0, stream>>>(adjm, hb, g);
  gemm2_glu<<<(B_ * TM2) * (MP / 64),                   256, 0, stream>>>(g, wt, bias, out);
}
